// TimeAwareMultiHeadAttention_9552007266735
// MI455X (gfx1250) — matrix operand order repaired and checked by reading, not compiled, not run on hardware
//
#include <hip/hip_runtime.h>
#include <hip/hip_bf16.h>

typedef __attribute__((ext_vector_type(16))) _Float16 v16h;
typedef __attribute__((ext_vector_type(8)))  float    v8f;
typedef __attribute__((ext_vector_type(8)))  unsigned int v8u;
typedef int vec4i __attribute__((vector_size(16)));  // matches builtin's V4i param

#define D_MODEL   1024
#define NUM_HEADS 16
#define D_K       64
#define BATCH     4
#define SEQ       1024
#define ROWS      (BATCH * SEQ)   // 4096

struct alignas(8) H4 { _Float16 h[4]; };

// ---------------------------------------------------------------------------
// CDNA5 async global->LDS staging (ASYNCcnt path), with portable fallback.
// ---------------------------------------------------------------------------
#if __has_builtin(__builtin_amdgcn_global_load_async_to_lds_b128)
#define HAS_ASYNC_LDS 1
#else
#define HAS_ASYNC_LDS 0
#endif

__device__ __forceinline__ void async_cp16(const _Float16* src, _Float16* dst) {
#if HAS_ASYNC_LDS
  __builtin_amdgcn_global_load_async_to_lds_b128(
      (__attribute__((address_space(1))) vec4i*)(unsigned long long)(uintptr_t)src,
      (__attribute__((address_space(3))) vec4i*)(unsigned int)(uintptr_t)dst,
      0, 0);
#else
  *(uint4*)dst = *(const uint4*)src;
#endif
}

__device__ __forceinline__ void async_fence() {
#if HAS_ASYNC_LDS
#if __has_builtin(__builtin_amdgcn_s_wait_asynccnt)
  __builtin_amdgcn_s_wait_asynccnt(0);
#else
  asm volatile("s_wait_asynccnt 0x0" ::: "memory");
#endif
#endif
}

// ---------------------------------------------------------------------------
// Fragment helpers (A/B: contiguous k order used by this code, same for both)
// A 16x32 f16: lane(half=l>>4, m=l&15); e -> K = half*16 + e
//   => per lane TWO contiguous 16B chunks at K = half*16 and K = half*16+8.
// B 32x16 f16: lane n=l&15; e -> K = half*16 + e  => 32 contiguous bytes.
// C/D 16x16 f32: reg r -> M = r + 8*half, N = l&15
// ---------------------------------------------------------------------------
__device__ __forceinline__ v16h pack16(uint4 x, uint4 y) {
  v8u r;
  r[0] = x.x; r[1] = x.y; r[2] = x.z; r[3] = x.w;
  r[4] = y.x; r[5] = y.y; r[6] = y.z; r[7] = y.w;
  return __builtin_bit_cast(v16h, r);
}

// row must be 16B-aligned; koff in {0,32}
__device__ __forceinline__ v16h load_afrag(const _Float16* row, int lhalf, int koff) {
  uint4 x = *(const uint4*)(row + koff + lhalf * 16);
  uint4 y = *(const uint4*)(row + koff + lhalf * 16 + 8);
  return pack16(x, y);
}

// p must be 16B-aligned (32 contiguous bytes)
__device__ __forceinline__ v16h load_bfrag(const _Float16* p) {
  const uint4* q = (const uint4*)p;
  return pack16(q[0], q[1]);
}

// ---------------------------------------------------------------------------
// Kernel 0: sum(Wt), sum(bt)
// ---------------------------------------------------------------------------
__global__ void reduce_wt_kernel(const float* __restrict__ Wt,
                                 const float* __restrict__ bt,
                                 float* __restrict__ sums) {
  __shared__ float sw[D_K], sb[D_K];
  int l = threadIdx.x;
  sw[l] = Wt[l];
  sb[l] = bt[l];
  __syncthreads();
  if (l == 0) {
    float a = 0.f, c = 0.f;
    for (int i = 0; i < D_K; ++i) { a += sw[i]; c += sb[i]; }
    sums[0] = a;
    sums[1] = c;
  }
}

// ---------------------------------------------------------------------------
// Kernel 1: f32 -> f16 streaming convert (activations)
// ---------------------------------------------------------------------------
__global__ __launch_bounds__(256) void cvt_act_kernel(
    const float* __restrict__ src, _Float16* __restrict__ dst, int n4) {
  int i = blockIdx.x * 256 + threadIdx.x;
  if (i < n4) {
    float4 f = ((const float4*)src)[i];
    H4 v;
    v.h[0] = (_Float16)f.x; v.h[1] = (_Float16)f.y;
    v.h[2] = (_Float16)f.z; v.h[3] = (_Float16)f.w;
    ((H4*)dst)[i] = v;
  }
}

// ---------------------------------------------------------------------------
// Kernel 2: weight convert + transpose: WT[n][k] = (f16) W[k][n]
// ---------------------------------------------------------------------------
__global__ __launch_bounds__(256) void cvt_wt_kernel(
    const float* __restrict__ W, _Float16* __restrict__ WT) {
  __shared__ _Float16 T[64][72];  // 144B rows -> uint4-aligned
  const int tid = threadIdx.x;
  const int kBase = blockIdx.x * 64;
  const int nBase = blockIdx.y * 64;
#pragma unroll
  for (int i = 0; i < 4; ++i) {
    int idx = tid + i * 256;            // 64 rows x 16 float4
    int kr = idx >> 4, c4 = idx & 15;
    float4 f = *(const float4*)(W + (size_t)(kBase + kr) * D_MODEL + nBase + c4 * 4);
    T[c4 * 4 + 0][kr] = (_Float16)f.x;
    T[c4 * 4 + 1][kr] = (_Float16)f.y;
    T[c4 * 4 + 2][kr] = (_Float16)f.z;
    T[c4 * 4 + 3][kr] = (_Float16)f.w;
  }
  __syncthreads();
#pragma unroll
  for (int i = 0; i < 2; ++i) {
    int idx = tid + i * 256;            // 64 rows x 8 granules of 8 f16
    int r = idx >> 3, c = idx & 7;
    *(uint4*)(WT + (size_t)(nBase + r) * D_MODEL + kBase + c * 8) =
        *(const uint4*)&T[r][c * 8];
  }
}

// ---------------------------------------------------------------------------
// Tiled WMMA GEMM with double-buffered async staging:
// Y[4096,1024] = A16 @ WT^T + bias. Block tile 128x64, K-chunk 64, 256 thr.
// Pipeline: async-stage tile(i+1) -> WMMA on tile(i) -> wait_async -> barrier.
// ---------------------------------------------------------------------------
template <bool HEADSPLIT>
__global__ __launch_bounds__(256) void gemm1024_kernel(
    const _Float16* __restrict__ A,
    const _Float16* __restrict__ WT,
    const float* __restrict__ bias,
    void* __restrict__ Out) {
  constexpr int BM = 128, BN = 64, BK = 64;
  constexpr int PAD = BK + 8;  // 72 f16 -> 144B rows, uint4-aligned
  constexpr int NIT = D_MODEL / BK;  // 16

  __shared__ _Float16 As[2][BM][PAD];
  __shared__ _Float16 Bs[2][BN][PAD];  // Bs[n][k]

  const int tid   = threadIdx.x;
  const int wave  = tid >> 5;
  const int lane  = tid & 31;
  const int lhalf = lane >> 4;
  const int ln    = lane & 15;

  const int rowBase = blockIdx.y * BM;
  const int nBase   = blockIdx.x * BN;

  auto stage = [&](int buf, int k0) {
#pragma unroll
    for (int i = 0; i < 4; ++i) {
      int idx = tid + i * 256;            // 1024 granules (8 f16), 8 per row
      int r = idx >> 3, c = idx & 7;
      async_cp16(A + (size_t)(rowBase + r) * D_MODEL + k0 + c * 8,
                 &As[buf][r][c * 8]);
    }
#pragma unroll
    for (int i = 0; i < 2; ++i) {
      int idx = tid + i * 256;            // 512 granules
      int r = idx >> 3, c = idx & 7;
      async_cp16(WT + (size_t)(nBase + r) * D_MODEL + k0 + c * 8,
                 &Bs[buf][r][c * 8]);
    }
  };

  const v8f zero = {0.f, 0.f, 0.f, 0.f, 0.f, 0.f, 0.f, 0.f};
  v8f acc[4];
#pragma unroll
  for (int t = 0; t < 4; ++t) acc[t] = zero;

  stage(0, 0);
  async_fence();
  __syncthreads();

  for (int it = 0; it < NIT; ++it) {
    const int buf = it & 1;
    if (it + 1 < NIT) stage(buf ^ 1, (it + 1) * BK);  // overlap with compute

    const _Float16* arow = &As[buf][wave * 16 + ln][0];
    v16h a0 = load_afrag(arow, lhalf, 0);
    v16h a1 = load_afrag(arow, lhalf, 32);
#pragma unroll
    for (int t = 0; t < 4; ++t) {
      const _Float16* brow = &Bs[buf][t * 16 + ln][0];
      v16h b0 = load_bfrag(brow + lhalf * 16);
      acc[t] = __builtin_amdgcn_wmma_f32_16x16x32_f16(
          false, a0, false, b0, (short)0, acc[t], false, false);
      v16h b1 = load_bfrag(brow + 32 + lhalf * 16);
      acc[t] = __builtin_amdgcn_wmma_f32_16x16x32_f16(
          false, a1, false, b1, (short)0, acc[t], false, false);
    }
    async_fence();
    __syncthreads();
  }

  // ---- epilogue: bias + store ----
#pragma unroll
  for (int t = 0; t < 4; ++t) {
    int col = nBase + t * 16 + ln;
    float bcol = bias[col];
#pragma unroll
    for (int r = 0; r < 8; ++r) {
      int row = rowBase + wave * 16 + r + 8 * lhalf;
      float val = acc[t][r] + bcol;
      if (HEADSPLIT) {
        int h = col >> 6, dk = col & 63;
        int bb = row >> 10, s = row & (SEQ - 1);
        ((_Float16*)Out)[((size_t)(bb * NUM_HEADS + h) * SEQ + s) * D_K + dk] =
            (_Float16)val;
      } else {
        ((float*)Out)[(size_t)row * D_MODEL + col] = val;
      }
    }
  }
}

// ---------------------------------------------------------------------------
// Flash-style attention per (b,h), double-buffered K/V tiles:
//   - K tile (kt+1) async-staged into LDS while computing tile kt
//   - V tile (kt+1) global loads issued before compute, swizzle-stored after
// ---------------------------------------------------------------------------
__global__ __launch_bounds__(128) void attn_kernel(
    const _Float16* __restrict__ Qh, const _Float16* __restrict__ Kh,
    const _Float16* __restrict__ Vh, const float* __restrict__ time_diffs,
    const int* __restrict__ mask, const float* __restrict__ sums,
    _Float16* __restrict__ Oa) {
  constexpr int PAD = 72;  // 144B rows -> uint4-aligned
  __shared__ _Float16 Ks[2][64][PAD];    // K tile  [key][d]
  __shared__ _Float16 Vt[2][64][PAD];    // V tile transposed [d][key]
  __shared__ _Float16 Pw[4][16][PAD];    // per-wave P tile [m][key]

  const int tid   = threadIdx.x;
  const int wave  = tid >> 5;
  const int lane  = tid & 31;
  const int lhalf = lane >> 4;
  const int ln    = lane & 15;

  const int bh    = blockIdx.y;
  const int bb    = bh >> 4;
  const int h     = bh & 15;
  const int qtile = blockIdx.x * 64;
  const size_t headBase = (size_t)bh * SEQ * D_K;

  const float sumWt = sums[0];
  const float sumBt = sums[1];
  const float scale = 0.125f;  // 1/sqrt(64)

  auto stageK = [&](int buf, int kbase) {
#pragma unroll
    for (int i = 0; i < 4; ++i) {
      int idx = tid + i * 128;            // 512 16B granules
      int r = idx >> 3, c = idx & 7;
      async_cp16(Kh + headBase + (size_t)(kbase + r) * D_K + c * 8,
                 &Ks[buf][r][c * 8]);
    }
  };

  unsigned vreg[16];  // V tile in flight (pairs d,d+1 of one key row)
  auto loadV = [&](int kbase) {
#pragma unroll
    for (int i = 0; i < 16; ++i) {
      int idx = tid + i * 128;
      int r = idx >> 5, c = idx & 31;
      vreg[i] = *(const unsigned*)(Vh + headBase + (size_t)(kbase + r) * D_K + c * 2);
    }
  };
  auto storeV = [&](int buf) {
#pragma unroll
    for (int i = 0; i < 16; ++i) {
      int idx = tid + i * 128;
      int r = idx >> 5, c = idx & 31;
      union { unsigned u; _Float16 h[2]; } cv; cv.u = vreg[i];
      Vt[buf][c * 2 + 0][r] = cv.h[0];
      Vt[buf][c * 2 + 1][r] = cv.h[1];
    }
  };

  // Q strip fragments for this wave (16 rows x 64 d = 2 A fragments)
  v16h aq[2];
  {
    const _Float16* qrow =
        Qh + headBase + (size_t)(qtile + wave * 16 + ln) * D_K;
#pragma unroll
    for (int c = 0; c < 2; ++c) aq[c] = load_afrag(qrow, lhalf, c * 32);
  }

  const v8f zero = {0.f, 0.f, 0.f, 0.f, 0.f, 0.f, 0.f, 0.f};
  v8f o[4];
  float mrun[8], lrun[8];
#pragma unroll
  for (int t = 0; t < 4; ++t) o[t] = zero;
#pragma unroll
  for (int r = 0; r < 8; ++r) { mrun[r] = -1e30f; lrun[r] = 0.f; }

  // prologue: tile 0
  stageK(0, 0);
  loadV(0);
  storeV(0);
  async_fence();
  __syncthreads();

  for (int kt = 0; kt < 16; ++kt) {
    const int kbase = kt * 64;
    const int buf = kt & 1;
    const bool more = (kt + 1) < 16;
    if (more) {
      stageK(buf ^ 1, kbase + 64);  // async, overlapped with compute
      loadV(kbase + 64);            // loads in flight through compute
    }

    // ---- scores = Q . K^T ----
    v8f s[4];
#pragma unroll
    for (int t = 0; t < 4; ++t) {
      s[t] = zero;
#pragma unroll
      for (int c = 0; c < 2; ++c) {
        v16h bk = load_bfrag(&Ks[buf][t * 16 + ln][c * 32 + lhalf * 16]);
        s[t] = __builtin_amdgcn_wmma_f32_16x16x32_f16(
            false, aq[c], false, bk, (short)0, s[t], false, false);
      }
    }

    // ---- scale, time decay, mask ----
    float sc[4][8];
#pragma unroll
    for (int t = 0; t < 4; ++t) {
#pragma unroll
      for (int r = 0; r < 8; ++r) {
        int qg = qtile + wave * 16 + r + 8 * lhalf;
        int kg = kbase + t * 16 + ln;
        size_t ix = ((size_t)bb * SEQ + qg) * SEQ + kg;
        float td = time_diffs[ix];
        int mk = mask[ix];
        float val = s[t][r] * scale - (td * sumWt + sumBt);
        sc[t][r] = (mk == 0) ? -1e9f : val;
      }
    }

    // ---- online softmax (row reductions across the 16-lane half) ----
#pragma unroll
    for (int r = 0; r < 8; ++r) {
      float rm = sc[0][r];
#pragma unroll
      for (int t = 1; t < 4; ++t) rm = fmaxf(rm, sc[t][r]);
#pragma unroll
      for (int off = 8; off >= 1; off >>= 1)
        rm = fmaxf(rm, __shfl_xor(rm, off, 32));
      float nm = fmaxf(mrun[r], rm);
      float alpha = __expf(mrun[r] - nm);
      mrun[r] = nm;
      float rs = 0.f;
#pragma unroll
      for (int t = 0; t < 4; ++t) {
        float p = __expf(sc[t][r] - nm);
        sc[t][r] = p;
        rs += p;
      }
#pragma unroll
      for (int off = 8; off >= 1; off >>= 1) rs += __shfl_xor(rs, off, 32);
      lrun[r] = lrun[r] * alpha + rs;
#pragma unroll
      for (int t = 0; t < 4; ++t) o[t][r] *= alpha;
    }

    // ---- P (C layout) -> wave-private LDS -> A fragments ----
#pragma unroll
    for (int t = 0; t < 4; ++t)
#pragma unroll
      for (int r = 0; r < 8; ++r)
        Pw[wave][r + 8 * lhalf][t * 16 + ln] = (_Float16)sc[t][r];
    asm volatile("s_wait_dscnt 0x0" ::: "memory");  // wave-local LDS RAW fence

    v16h ap[2];
#pragma unroll
    for (int c = 0; c < 2; ++c)
      ap[c] = load_afrag(&Pw[wave][ln][0], lhalf, c * 32);

    // ---- O += P @ V ----
#pragma unroll
    for (int t = 0; t < 4; ++t) {
#pragma unroll
      for (int c = 0; c < 2; ++c) {
        v16h bv = load_bfrag(&Vt[buf][t * 16 + ln][c * 32 + lhalf * 16]);
        o[t] = __builtin_amdgcn_wmma_f32_16x16x32_f16(
            false, ap[c], false, bv, (short)0, o[t], false, false);
      }
    }

    if (more) storeV(buf ^ 1);  // commit next V tile (loads have had compute to hide)
    async_fence();
    __syncthreads();
  }

  // ---- normalize + store f16 [B,S,D] ----
#pragma unroll
  for (int t = 0; t < 4; ++t) {
#pragma unroll
    for (int r = 0; r < 8; ++r) {
      int qg = qtile + wave * 16 + r + 8 * lhalf;
      int d = h * D_K + t * 16 + ln;
      float val = o[t][r] / lrun[r];
      Oa[((size_t)bb * SEQ + qg) * D_MODEL + d] = (_Float16)val;
    }
  }
}

// ---------------------------------------------------------------------------
// Launch
// ---------------------------------------------------------------------------
extern "C" void kernel_launch(void* const* d_in, const int* in_sizes, int n_in,
                              void* d_out, int out_size, void* d_ws, size_t ws_size,
                              hipStream_t stream) {
  const float* query = (const float*)d_in[0];
  const float* key   = (const float*)d_in[1];
  const float* value = (const float*)d_in[2];
  const float* tdif  = (const float*)d_in[3];
  const int*   mask  = (const int*)d_in[4];
  const float* Wq = (const float*)d_in[5];
  const float* bq = (const float*)d_in[6];
  const float* Wk = (const float*)d_in[7];
  const float* bk = (const float*)d_in[8];
  const float* Wv = (const float*)d_in[9];
  const float* bv = (const float*)d_in[10];
  const float* Wt = (const float*)d_in[11];
  const float* bt = (const float*)d_in[12];
  const float* Wo = (const float*)d_in[13];
  const float* bo = (const float*)d_in[14];

  const size_t mat  = (size_t)ROWS * D_MODEL;     // 4,194,304 elements
  const size_t wmat = (size_t)D_MODEL * D_MODEL;  // 1,048,576 elements
  _Float16* Qh  = (_Float16*)d_ws;
  _Float16* Kh  = Qh + mat;
  _Float16* Vh  = Kh + mat;
  _Float16* Ao  = Vh + mat;
  _Float16* A16 = Ao + mat;     // reused per-GEMM f16 activations
  _Float16* WT  = A16 + mat;    // reused per-GEMM transposed f16 weights
  float* sums   = (float*)(WT + wmat);

  reduce_wt_kernel<<<1, 64, 0, stream>>>(Wt, bt, sums);

  const int n4 = (int)(mat / 4);
  dim3 cgrid((n4 + 255) / 256);
  dim3 wgrid(D_MODEL / 64, D_MODEL / 64);  // (16,16)
  dim3 ggrid(D_MODEL / 64, ROWS / 128);    // (16,32)

  // Q projection
  cvt_act_kernel<<<cgrid, 256, 0, stream>>>(query, A16, n4);
  cvt_wt_kernel<<<wgrid, 256, 0, stream>>>(Wq, WT);
  gemm1024_kernel<true><<<ggrid, 256, 0, stream>>>(A16, WT, bq, Qh);
  // K projection
  cvt_act_kernel<<<cgrid, 256, 0, stream>>>(key, A16, n4);
  cvt_wt_kernel<<<wgrid, 256, 0, stream>>>(Wk, WT);
  gemm1024_kernel<true><<<ggrid, 256, 0, stream>>>(A16, WT, bk, Kh);
  // V projection
  cvt_act_kernel<<<cgrid, 256, 0, stream>>>(value, A16, n4);
  cvt_wt_kernel<<<wgrid, 256, 0, stream>>>(Wv, WT);
  gemm1024_kernel<true><<<ggrid, 256, 0, stream>>>(A16, WT, bv, Vh);

  // Attention
  dim3 agrid(SEQ / 64, BATCH * NUM_HEADS);  // (16, 64)
  attn_kernel<<<agrid, 128, 0, stream>>>(Qh, Kh, Vh, tdif, mask, sums, Ao);

  // Output projection
  cvt_wt_kernel<<<wgrid, 256, 0, stream>>>(Wo, WT);
  gemm1024_kernel<false><<<ggrid, 256, 0, stream>>>(Ao, WT, bo, d_out);
}
